// SAGEConv_53163105190231
// MI455X (gfx1250) — matrix operand order repaired and checked by reading, not compiled, not run on hardware
//
#include <hip/hip_runtime.h>
#include <hip/hip_bf16.h>

typedef __attribute__((ext_vector_type(16))) __bf16 v16bf;
typedef __attribute__((ext_vector_type(8)))  __bf16 v8bf;
typedef __attribute__((ext_vector_type(8)))  float  v8f;

#define DIM   256      // input feature dim
#define KDIM  512      // 2*DIM (concat [x, agg])
#define KNEI  16       // neighbors per node
#define OUTD  256      // output dim

// ---- bf16 helpers (round-to-nearest-even) -------------------------------
__device__ __forceinline__ __bf16 f2bf(float f) {
    unsigned u = __float_as_uint(f);
    unsigned r = (u + 0x7FFFu + ((u >> 16) & 1u)) >> 16;
    unsigned short s = (unsigned short)r;
    return __builtin_bit_cast(__bf16, s);
}
__device__ __forceinline__ float bf2f(__bf16 h) {
    unsigned short s = __builtin_bit_cast(unsigned short, h);
    return __uint_as_float(((unsigned)s) << 16);
}

// ---- CDNA5 async copy engine helpers ------------------------------------
// GLOBAL_LOAD_ASYNC_TO_LDS_B64 (opcode 97): per-lane 8B global -> LDS,
// tracked with ASYNCcnt; LDS dest offset comes from a VGPR (VDST slot).
__device__ __forceinline__ void async_g2l_b64(unsigned lds_off, const void* gaddr) {
    asm volatile("global_load_async_to_lds_b64 %0, %1, off"
                 :: "v"(lds_off), "v"(gaddr) : "memory");
}
__device__ __forceinline__ void wait_async0() {
#if __has_builtin(__builtin_amdgcn_s_wait_asynccnt)
    __builtin_amdgcn_s_wait_asynccnt(0);
#else
    asm volatile("s_wait_asynccnt 0x0" ::: "memory");
#endif
}

// ---- Kernel 0: split W (fp32) into hi/lo bf16 ---------------------------
__global__ void splitW_kernel(const float* __restrict__ W,
                              __bf16* __restrict__ Whi,
                              __bf16* __restrict__ Wlo, int n) {
    int i = blockIdx.x * blockDim.x + threadIdx.x;
    if (i < n) {
        float w  = W[i];
        __bf16 h = f2bf(w);
        Whi[i] = h;
        Wlo[i] = f2bf(w - bf2f(h));
    }
}

// ---- Kernel 1: gather-mean + pack h = [x | agg] as hi/lo bf16 -----------
__global__ __launch_bounds__(DIM)
void agg_pack_kernel(const float* __restrict__ x,
                     const int*   __restrict__ neigh,
                     __bf16* __restrict__ hHi,
                     __bf16* __restrict__ hLo, int N) {
    const int m = blockIdx.x;       // m = b*N + n
    const int n = m % N;
    const int d = threadIdx.x;

    __shared__ int nb[KNEI];
    if (threadIdx.x < KNEI) nb[threadIdx.x] = neigh[(size_t)n * KNEI + threadIdx.x];
    __syncthreads();

    const float* xb = x + (size_t)(m - n) * DIM;   // batch base: b*N*DIM

    float s = 0.f; int cnt = 0;
    #pragma unroll
    for (int k = 0; k < KNEI; ++k) {
        int j = nb[k];                              // wave-uniform branch
        if (j >= 0) { s += xb[(size_t)j * DIM + d]; ++cnt; }
    }
    float a  = s / (float)(cnt > 0 ? cnt : 1);
    float xv = xb[(size_t)n * DIM + d];

    size_t r = (size_t)m * KDIM;
    __bf16 xh = f2bf(xv);
    hHi[r + d] = xh;
    hLo[r + d] = f2bf(xv - bf2f(xh));
    __bf16 ah = f2bf(a);
    hHi[r + DIM + d] = ah;
    hLo[r + DIM + d] = f2bf(a - bf2f(ah));
}

// ---- Kernel 2: split-bf16 WMMA GEMM: out = relu(h @ W^T + b) ------------
// Block = 128 threads = 4 waves; one 16-row M-tile per block; wave w owns
// 64 output cols (NT=4 tiles). A-tile (16x32, hi+lo) is double-buffered in
// LDS and staged with the async copy engine so the next tile's fetch
// overlaps the current tile's WMMAs; one barrier + one s_wait_asynccnt per
// K-chunk.
__global__ __launch_bounds__(128)
void sage_gemm_kernel(const __bf16* __restrict__ hHi,
                      const __bf16* __restrict__ hLo,
                      const __bf16* __restrict__ Whi,
                      const __bf16* __restrict__ Wlo,
                      const float*  __restrict__ bias,
                      float* __restrict__ out, int M) {
    constexpr int NT = 4;                       // 16-col tiles per wave
    constexpr int KC = KDIM / 32;               // 16 K-chunks

    // [buffer][hi/lo][16 rows x 32 k]
    __shared__ __align__(16) __bf16 sA[2][2][16 * 32];

    const int tid  = threadIdx.x;
    const int lane = tid & 31;
    const int wave = tid >> 5;                  // 0..3 -> column group
    const int m0   = blockIdx.x * 16;
    const int n0   = wave * (NT * 16);

    // A fragment addressing (ISA 7.12.2, 16-bit A 16x32):
    //   lanes 0-15: M=lane, K={0..7,16..23}; lanes 16-31: M=lane-16, K={8..15,24..31}
    const int arow  = lane & 15;
    const int akoff = (lane < 16) ? 0 : 8;
    // B fragment (32x16): lanes 0-15 K={0..7,16..23}, lanes 16-31 K={8..15,24..31}, N=lane%16
    const int bkoff = (lane < 16) ? 0 : 8;

    // Cooperative A-stage mapping: 512 elems / 128 threads = 4 bf16 (8B) each
    const int e    = tid * 4;
    const int grow = e >> 5;
    const int gk   = e & 31;

    // LDS byte offsets for this thread's async destinations (addr[31:0] of
    // the generic shared pointer is the workgroup-relative LDS offset).
    unsigned ldsHi[2], ldsLo[2];
    ldsHi[0] = (unsigned)(size_t)&sA[0][0][e];
    ldsLo[0] = (unsigned)(size_t)&sA[0][1][e];
    ldsHi[1] = (unsigned)(size_t)&sA[1][0][e];
    ldsLo[1] = (unsigned)(size_t)&sA[1][1][e];

    const size_t grow_base = (size_t)(m0 + grow) * KDIM + gk;
    auto stage = [&](int kc, int buf) {
        size_t gbase = grow_base + (size_t)kc * 32;
        async_g2l_b64(ldsHi[buf], &hHi[gbase]);
        async_g2l_b64(ldsLo[buf], &hLo[gbase]);
    };

    v8f acc[NT] = {};

    stage(0, 0);                                // prologue: fill buffer 0

    for (int kc = 0; kc < KC; ++kc) {
        const int cur = kc & 1;
        wait_async0();                          // own async copies landed
        __syncthreads();                        // everyone's copies visible;
                                                // buf[cur^1] free to overwrite
        if (kc + 1 < KC) stage(kc + 1, cur ^ 1);

        // ---- A fragments from LDS (two 16B ds loads each) ----
        const __bf16* pa = &sA[cur][0][arow * 32 + akoff];
        v8bf a0 = *reinterpret_cast<const v8bf*>(pa);
        v8bf a1 = *reinterpret_cast<const v8bf*>(pa + 16);
        v16bf aHi = __builtin_shufflevector(a0, a1, 0,1,2,3,4,5,6,7,8,9,10,11,12,13,14,15);
        const __bf16* pl = &sA[cur][1][arow * 32 + akoff];
        v8bf l0 = *reinterpret_cast<const v8bf*>(pl);
        v8bf l1 = *reinterpret_cast<const v8bf*>(pl + 16);
        v16bf aLo = __builtin_shufflevector(l0, l1, 0,1,2,3,4,5,6,7,8,9,10,11,12,13,14,15);

        #pragma unroll
        for (int t = 0; t < NT; ++t) {
            size_t wb = (size_t)(n0 + t * 16 + (lane & 15)) * KDIM + (size_t)kc * 32 + bkoff;
            v8bf b0 = *reinterpret_cast<const v8bf*>(&Whi[wb]);
            v8bf b1 = *reinterpret_cast<const v8bf*>(&Whi[wb + 16]);
            v16bf bHi = __builtin_shufflevector(b0, b1, 0,1,2,3,4,5,6,7,8,9,10,11,12,13,14,15);
            v8bf c0 = *reinterpret_cast<const v8bf*>(&Wlo[wb]);
            v8bf c1 = *reinterpret_cast<const v8bf*>(&Wlo[wb + 16]);
            v16bf bLo = __builtin_shufflevector(c0, c1, 0,1,2,3,4,5,6,7,8,9,10,11,12,13,14,15);

            // split-bf16 product: hi*HI + hi*LO + lo*HI (lo*LO ~ 2^-32, dropped)
            acc[t] = __builtin_amdgcn_wmma_f32_16x16x32_bf16(false, aHi, false, bHi,
                                                             (short)0, acc[t], false, false);
            acc[t] = __builtin_amdgcn_wmma_f32_16x16x32_bf16(false, aHi, false, bLo,
                                                             (short)0, acc[t], false, false);
            acc[t] = __builtin_amdgcn_wmma_f32_16x16x32_bf16(false, aLo, false, bHi,
                                                             (short)0, acc[t], false, false);
        }
    }

    // ---- epilogue: bias + relu; C layout: VGPR r -> M = r (+8 for lanes>=16)
    const int rofs = (lane < 16) ? 0 : 8;
    #pragma unroll
    for (int t = 0; t < NT; ++t) {
        int col  = n0 + t * 16 + (lane & 15);
        float bv = bias[col];
        #pragma unroll
        for (int r = 0; r < 8; ++r) {
            int row = m0 + r + rofs;
            float v = acc[t][r] + bv;
            v = v > 0.f ? v : 0.f;
            if (row < M) out[(size_t)row * OUTD + col] = v;
        }
    }
}

// ---- host launcher ------------------------------------------------------
extern "C" void kernel_launch(void* const* d_in, const int* in_sizes, int n_in,
                              void* d_out, int out_size, void* d_ws, size_t ws_size,
                              hipStream_t stream) {
    const float* x     = (const float*)d_in[0];  // (B,N,256) fp32
    const int*   neigh = (const int*)d_in[1];    // (N,16) int
    const float* W     = (const float*)d_in[2];  // (256,512) fp32
    const float* bias  = (const float*)d_in[3];  // (256,) fp32
    float*       out   = (float*)d_out;          // (B,N,256) fp32

    const int N  = in_sizes[1] / KNEI;           // 10000
    const int B  = in_sizes[0] / (N * DIM);      // 4
    const int M  = B * N;                        // 40000 rows
    const int Mt = (M + 15) / 16;                // M-tiles (M is 16-aligned here)

    // workspace layout: hHi | hLo | Whi | Wlo (bf16)
    size_t hElems = (size_t)Mt * 16 * KDIM;
    __bf16* hHi = (__bf16*)d_ws;
    __bf16* hLo = hHi + hElems;
    __bf16* Whi = hLo + hElems;
    __bf16* Wlo = Whi + (size_t)OUTD * KDIM;

    splitW_kernel<<<(OUTD * KDIM + 255) / 256, 256, 0, stream>>>(W, Whi, Wlo, OUTD * KDIM);
    agg_pack_kernel<<<M, DIM, 0, stream>>>(x, neigh, hHi, hLo, N);
    sage_gemm_kernel<<<Mt, 128, 0, stream>>>(hHi, hLo, Whi, Wlo, bias, out, M);
}
